// MultiHeadAttention_40750649704762
// MI455X (gfx1250) — matrix operand order repaired and checked by reading, not compiled, not run on hardware
//
#include <hip/hip_runtime.h>
#include <hip/hip_bf16.h>
#include <stdint.h>

typedef __attribute__((ext_vector_type(16))) __bf16 v16bf;
typedef __attribute__((ext_vector_type(8)))  __bf16 v8bf;
typedef __attribute__((ext_vector_type(8)))  float  v8f;

#define D_MODEL 1024
#define SEQ     2048
#define BATCH   4
#define NHEAD   16
#define HDIM    64

__device__ __forceinline__ unsigned short f2bfu(float f) {
  unsigned u = __float_as_uint(f);
  u += 0x7FFFu + ((u >> 16) & 1u);   // round-to-nearest-even
  return (unsigned short)(u >> 16);
}

// ---------------- fp32 -> bf16 conversion (4 elems / thread) ----------------
__global__ __launch_bounds__(256) void cvt_f32_bf16(const float* __restrict__ in,
                                                    unsigned short* __restrict__ out,
                                                    int n4) {
  int i = blockIdx.x * 256 + threadIdx.x;
  if (i >= n4) return;
  float4 a = reinterpret_cast<const float4*>(in)[i];
  ushort4 r;
  r.x = f2bfu(a.x); r.y = f2bfu(a.y); r.z = f2bfu(a.z); r.w = f2bfu(a.w);
  reinterpret_cast<ushort4*>(out)[i] = r;
}

// ---------------- NT GEMM: C[M,N] = A[M,K] * W[N,K]^T  (bf16 -> f32 acc) ----
// Block: 256 threads = 8 waves; block tile 128(M) x 128(N); K-step 32.
// Each wave computes a 16x128 strip (8 f32 accumulators = 64 VGPRs).
// Double-buffered LDS B tile (2 x 8KB), ONE barrier per K-step:
//   iter i: [preload W tile i+1 -> regs] [8 WMMAs from Bs[p]] [regs -> Bs[p^1]] [sync]
// OUTMODE 0: f32 out row-major [M,N] + bias     (final projection)
// OUTMODE 1: bf16 out in [B,H,T,d] head layout  (Q, K)
// OUTMODE 2: bf16 out in [B,H,d,T] (transposed) (V)
template <int OUTMODE>
__global__ __launch_bounds__(256) void gemm_nt_bf16(
    const unsigned short* __restrict__ A,
    const unsigned short* __restrict__ W,
    float* __restrict__ outF,
    unsigned short* __restrict__ outB,
    const float* __restrict__ bias,
    int M, int N, int K)
{
  __shared__ __align__(32) unsigned short Bs[2][128 * 32];  // 2 x 8KB
  const int tid  = threadIdx.x;
  const int wave = tid >> 5;
  const int lane = tid & 31;
  const int half = lane >> 4;
  const int lr   = lane & 15;
  const int m0 = blockIdx.x * 128 + wave * 16;
  const int n0 = blockIdx.y * 128;
  const unsigned short* Arow = A + (size_t)(m0 + lr) * K;
  const int sr = tid >> 1;          // staging row   0..127
  const int sc = (tid & 1) * 16;    // staging chunk 0 or 16
  const unsigned short* Wst = W + (size_t)(n0 + sr) * K + sc;

  v8f acc[8] = {};
  // prologue: stage K-tile 0 into buffer 0
  {
    v16bf w0 = *reinterpret_cast<const v16bf*>(Wst);
    *reinterpret_cast<v16bf*>(&Bs[0][sr * 32 + sc]) = w0;
  }
  __syncthreads();

  const int niter = K / 32;
  for (int i = 0; i < niter; ++i) {
    const int k0 = i * 32;
    const int p  = i & 1;
    // preload next W tile into registers (overlaps with WMMA below)
    v16bf wreg;
    if (i + 1 < niter)
      wreg = *reinterpret_cast<const v16bf*>(Wst + (i + 1) * 32);
    // prefetch A stream two K-steps ahead (global_prefetch_b8)
    if (i + 2 < niter)
      __builtin_prefetch(Arow + k0 + 64, 0, 0);
    // A fragment: row = lr, contiguous K chunk at 16*half
    v8bf a0 = *reinterpret_cast<const v8bf*>(Arow + k0 + 16 * half);
    v8bf a1 = *reinterpret_cast<const v8bf*>(Arow + k0 + 16 * half + 8);
    v16bf a = __builtin_shufflevector(a0, a1, 0,1,2,3,4,5,6,7,8,9,10,11,12,13,14,15);
#pragma unroll
    for (int nt = 0; nt < 8; ++nt) {
      // B fragment: col = lr, contiguous K chunk at 16*half
      v16bf b = *reinterpret_cast<const v16bf*>(&Bs[p][(nt * 16 + lr) * 32 + 16 * half]);
      acc[nt] = __builtin_amdgcn_wmma_f32_16x16x32_bf16(
          false, a, false, b, (short)0, acc[nt], false, false);
    }
    if (i + 1 < niter)
      *reinterpret_cast<v16bf*>(&Bs[p ^ 1][sr * 32 + sc]) = wreg;
    __syncthreads();
  }

#pragma unroll
  for (int nt = 0; nt < 8; ++nt) {
#pragma unroll
    for (int v = 0; v < 8; ++v) {
      int m = m0 + v + 8 * half;          // C layout: row = vgpr + 8*half
      int n = n0 + nt * 16 + lr;          //           col = lane & 15
      float val = acc[nt][v];
      if (OUTMODE == 0) {
        outF[(size_t)m * N + n] = val + bias[n];
      } else {
        int bb = m >> 11, t = m & 2047, hh = n >> 6, dc = n & 63;
        if (OUTMODE == 1)
          outB[(((size_t)(bb * NHEAD + hh) * SEQ + t) << 6) + dc] = f2bfu(val);
        else
          outB[((size_t)(bb * NHEAD + hh) * HDIM + dc) * SEQ + t] = f2bfu(val);
      }
    }
  }
}

// ---------------- fused causal flash attention (bf16 WMMA, f32 online softmax)
// grid (T/64, H, B); block 128 = 4 waves; each wave: 16 q-rows.
__global__ __launch_bounds__(128) void attn_fused(
    const unsigned short* __restrict__ Qb,   // [B,H,T,d] bf16
    const unsigned short* __restrict__ Kb,   // [B,H,T,d] bf16
    const unsigned short* __restrict__ Vtb,  // [B,H,d,T] bf16
    unsigned short* __restrict__ ctxb)       // [B,T,D]   bf16
{
  __shared__ __align__(32) unsigned short P[4][16 * 64];  // per-wave P tile
  const int qt   = blockIdx.x;
  const int h    = blockIdx.y;
  const int b    = blockIdx.z;
  const int wave = threadIdx.x >> 5;
  const int lane = threadIdx.x & 31;
  const int half = lane >> 4;
  const int lr   = lane & 15;
  const size_t bh = (size_t)(b * NHEAD + h);
  const unsigned short* Q  = Qb  + bh * (SEQ * HDIM);
  const unsigned short* Kp = Kb  + bh * (SEQ * HDIM);
  const unsigned short* Vt = Vtb + bh * (HDIM * SEQ);
  const int q0 = qt * 64 + wave * 16;

  // Q fragments for the two 32-wide d-steps (resident for whole kv loop)
  v16bf qf[2];
  {
    const unsigned short* qp = Q + (size_t)(q0 + lr) * HDIM;
#pragma unroll
    for (int s = 0; s < 2; ++s) {
      v8bf x0 = *reinterpret_cast<const v8bf*>(qp + s * 32 + 16 * half);
      v8bf x1 = *reinterpret_cast<const v8bf*>(qp + s * 32 + 16 * half + 8);
      qf[s] = __builtin_shufflevector(x0, x1, 0,1,2,3,4,5,6,7,8,9,10,11,12,13,14,15);
    }
  }

  v8f o[4] = {};
  float mrow[8], lrow[8];
#pragma unroll
  for (int v = 0; v < 8; ++v) { mrow[v] = -3.0e38f; lrow[v] = 0.0f; }
  unsigned short* Pw = P[wave];

  for (int kt = 0; kt <= qt; ++kt) {
    const int j0 = kt * 64;
    // ---- S = Q * K^T  (16 x 64 tile as 4 frags) ----
    v8f s4[4] = {};
#pragma unroll
    for (int jt = 0; jt < 4; ++jt) {
#pragma unroll
      for (int ds = 0; ds < 2; ++ds) {
        v16bf kb = *reinterpret_cast<const v16bf*>(
            Kp + (size_t)(j0 + jt * 16 + lr) * HDIM + ds * 32 + 16 * half);
        s4[jt] = __builtin_amdgcn_wmma_f32_16x16x32_bf16(
            false, qf[ds], false, kb, (short)0, s4[jt], false, false);
      }
    }
    // ---- scale 1/sqrt(64), causal mask on diagonal tile ----
    const bool diag = (kt == qt);
#pragma unroll
    for (int jt = 0; jt < 4; ++jt) {
      const int col = j0 + jt * 16 + lr;
#pragma unroll
      for (int v = 0; v < 8; ++v) {
        float s = s4[jt][v] * 0.125f;
        if (diag && col > (q0 + v + 8 * half)) s = -3.0e38f;
        s4[jt][v] = s;
      }
    }
    // ---- row max (reduce across 16-lane half via shfl_xor 8,4,2,1) ----
    float rm[8];
#pragma unroll
    for (int v = 0; v < 8; ++v)
      rm[v] = fmaxf(fmaxf(s4[0][v], s4[1][v]), fmaxf(s4[2][v], s4[3][v]));
#pragma unroll
    for (int off = 8; off >= 1; off >>= 1)
#pragma unroll
      for (int v = 0; v < 8; ++v) rm[v] = fmaxf(rm[v], __shfl_xor(rm[v], off, 32));
    // ---- online rescale ----
    float rs[8];
#pragma unroll
    for (int v = 0; v < 8; ++v) {
      float nm = fmaxf(mrow[v], rm[v]);
      float corr = __expf(mrow[v] - nm);
      mrow[v] = nm;
      lrow[v] *= corr;
#pragma unroll
      for (int dt = 0; dt < 4; ++dt) o[dt][v] *= corr;
      rs[v] = 0.0f;
    }
    // ---- P = exp(S - m), partial row sums ----
#pragma unroll
    for (int jt = 0; jt < 4; ++jt)
#pragma unroll
      for (int v = 0; v < 8; ++v) {
        float p = __expf(s4[jt][v] - mrow[v]);
        s4[jt][v] = p;
        rs[v] += p;
      }
#pragma unroll
    for (int off = 8; off >= 1; off >>= 1)
#pragma unroll
      for (int v = 0; v < 8; ++v) rs[v] += __shfl_xor(rs[v], off, 32);
#pragma unroll
    for (int v = 0; v < 8; ++v) lrow[v] += rs[v];
    // ---- C-frag -> A-frag relayout through per-wave LDS tile ----
#pragma unroll
    for (int jt = 0; jt < 4; ++jt)
#pragma unroll
      for (int v = 0; v < 8; ++v)
        Pw[(v + 8 * half) * 64 + jt * 16 + lr] = f2bfu(s4[jt][v]);
    asm volatile("s_wait_dscnt 0" ::: "memory");
    // ---- O += P * V  (V pre-transposed: B-operand chunks contiguous) ----
#pragma unroll
    for (int ds = 0; ds < 2; ++ds) {
      v8bf p0 = *reinterpret_cast<const v8bf*>(Pw + lr * 64 + ds * 32 + 16 * half);
      v8bf p1 = *reinterpret_cast<const v8bf*>(Pw + lr * 64 + ds * 32 + 16 * half + 8);
      v16bf pa = __builtin_shufflevector(p0, p1, 0,1,2,3,4,5,6,7,8,9,10,11,12,13,14,15);
#pragma unroll
      for (int dt = 0; dt < 4; ++dt) {
        v16bf vb = *reinterpret_cast<const v16bf*>(
            Vt + (size_t)(dt * 16 + lr) * SEQ + j0 + ds * 32 + 16 * half);
        o[dt] = __builtin_amdgcn_wmma_f32_16x16x32_bf16(
            false, pa, false, vb, (short)0, o[dt], false, false);
      }
    }
  }
  // ---- epilogue: ctx[b,t, h*64+dc] = O / l  (bf16 for final GEMM) ----
#pragma unroll
  for (int dt = 0; dt < 4; ++dt)
#pragma unroll
    for (int v = 0; v < 8; ++v) {
      int row = q0 + v + 8 * half;
      int dc  = dt * 16 + lr;
      float val = o[dt][v] / lrow[v];
      ctxb[((size_t)(b * SEQ + row)) * D_MODEL + h * HDIM + dc] = f2bfu(val);
    }
}

// ---------------------------------------------------------------------------
extern "C" void kernel_launch(void* const* d_in, const int* in_sizes, int n_in,
                              void* d_out, int out_size, void* d_ws, size_t ws_size,
                              hipStream_t stream) {
  const float* x   = (const float*)d_in[0];
  const float* w_q = (const float*)d_in[1];
  const float* w_k = (const float*)d_in[2];
  const float* w_v = (const float*)d_in[3];
  const float* w_o = (const float*)d_in[4];
  const float* b_o = (const float*)d_in[5];
  float* out = (float*)d_out;

  char* ws = (char*)d_ws;
  const size_t MB = 1024 * 1024;
  unsigned short* xb   = (unsigned short*)(ws +  0 * MB);  // 16 MB
  unsigned short* wqb  = (unsigned short*)(ws + 16 * MB);  //  2 MB
  unsigned short* wkb  = (unsigned short*)(ws + 18 * MB);
  unsigned short* wvb  = (unsigned short*)(ws + 20 * MB);
  unsigned short* wob  = (unsigned short*)(ws + 22 * MB);
  unsigned short* qb   = (unsigned short*)(ws + 24 * MB);  // 16 MB [B,H,T,d]
  unsigned short* kb   = (unsigned short*)(ws + 40 * MB);  // 16 MB [B,H,T,d]
  unsigned short* vtb  = (unsigned short*)(ws + 56 * MB);  // 16 MB [B,H,d,T]
  unsigned short* ctxb = (unsigned short*)(ws + 72 * MB);  // 16 MB [B,T,D]

  const int M = BATCH * SEQ;  // 8192
  // fp32 -> bf16
  cvt_f32_bf16<<<(M * D_MODEL / 4) / 256, 256, 0, stream>>>(x, xb, M * D_MODEL / 4);
  cvt_f32_bf16<<<(D_MODEL * D_MODEL / 4) / 256, 256, 0, stream>>>(w_q, wqb, D_MODEL * D_MODEL / 4);
  cvt_f32_bf16<<<(D_MODEL * D_MODEL / 4) / 256, 256, 0, stream>>>(w_k, wkb, D_MODEL * D_MODEL / 4);
  cvt_f32_bf16<<<(D_MODEL * D_MODEL / 4) / 256, 256, 0, stream>>>(w_v, wvb, D_MODEL * D_MODEL / 4);
  cvt_f32_bf16<<<(D_MODEL * D_MODEL / 4) / 256, 256, 0, stream>>>(w_o, wob, D_MODEL * D_MODEL / 4);

  dim3 g(M / 128, D_MODEL / 128);   // (64, 8)
  gemm_nt_bf16<1><<<g, 256, 0, stream>>>(xb, wqb, nullptr, qb,  nullptr, M, D_MODEL, D_MODEL);
  gemm_nt_bf16<1><<<g, 256, 0, stream>>>(xb, wkb, nullptr, kb,  nullptr, M, D_MODEL, D_MODEL);
  gemm_nt_bf16<2><<<g, 256, 0, stream>>>(xb, wvb, nullptr, vtb, nullptr, M, D_MODEL, D_MODEL);

  attn_fused<<<dim3(SEQ / 64, NHEAD, BATCH), 128, 0, stream>>>(qb, kb, vtb, ctxb);

  gemm_nt_bf16<0><<<g, 256, 0, stream>>>(ctxb, wob, out, nullptr, b_o, M, D_MODEL, D_MODEL);
}
